// VQVAE_43319040147479
// MI455X (gfx1250) — matrix operand order repaired and checked by reading, not compiled, not run on hardware
//
#include <hip/hip_runtime.h>
#include <math.h>

typedef __attribute__((ext_vector_type(16))) _Float16 v16h;
typedef __attribute__((ext_vector_type(8)))  float    v8f;
typedef __attribute__((ext_vector_type(4)))  unsigned v4u;
typedef __attribute__((ext_vector_type(8)))  int      v8i;
typedef __attribute__((ext_vector_type(4)))  int      v4i;

__device__ __forceinline__ unsigned pack2h(float a, float b)
{
    union { _Float16 h[2]; unsigned u; } p;
    p.h[0] = (_Float16)a; p.h[1] = (_Float16)b;
    return p.u;
}

// ---------------------------------------------------------------------------
// WMMA GEMM:  C[M,N] = epi( A[M,K] * B[K,N] + bias[M] )
//   A addressed as A[m*sa_m + k*sa_k] (lets us feed X^T for the VQ GEMM).
//   B row-major [K,N] contiguous.
//   EPI: 0 = none, 1 = exact GELU, 2 = add residual R[M,N]
//
// Block = 128 threads = 4 waves, block tile 16(M) x 128(N); each wave owns a
// 16x32 slab -> 2 WMMAs per k-step sharing one A fragment. A/B staged in LDS
// as f16 in fragment order (ISA 7.12.2): per-lane fetch = one 32B LDS read.
// Double-buffered LDS -> one barrier per k-step; interior tiles use uniform
// unconditional vector loads + packed f16-pair LDS stores.
// Fragment order: A (16m x 32k): lane = m + 16*g, g=(k>>3)&1,
//   idx = ((k&16)>>1)+(k&7).  B (32k x 16n): lane = n + 16*g, same idx.
// ---------------------------------------------------------------------------
template<int EPI>
__global__ __launch_bounds__(128)
void gemm_wmma_k(const float* __restrict__ A, long sa_m, long sa_k,
                 const float* __restrict__ B,
                 const float* __restrict__ bias,
                 const float* __restrict__ R,
                 float* __restrict__ Cmat,
                 int M, int N, int K)
{
    __shared__ __align__(32) unsigned AsU[2][32][8];        // A tile, fragment order
    __shared__ __align__(32) unsigned BsU[2][4][2][32][8];  // per-wave, 2 n-subtiles

    const int lane = threadIdx.x & 31;
    const int wv   = threadIdx.x >> 5;
    const int tm   = blockIdx.y * 16;
    const int tn   = blockIdx.x * 128 + wv * 32;
    const int half = lane >> 4;
    const int lm   = lane & 15;

    const bool a_full_m = (tm + 16 <= M);
    const bool b_full_n = (tn + 32 <= N);
    const bool b_vec    = ((N & 3) == 0);
    const bool a_colvec = (sa_m == 1) && ((sa_k & 3) == 0);  // column-major A, vec4-able

    auto stage = [&](int p, int k0) {
        // ---------------- A tile (16m x 32k) ----------------
        {
            const bool a_int = a_full_m && (k0 + 32 <= K);
            if (a_int && sa_k == 1) {
                // row-major A: thread -> 4 consecutive k at fixed m
                int j  = threadIdx.x * 4;
                int m  = j >> 5;
                int kb = j & 31;
                float4 va = *(const float4*)&A[(long)(tm + m) * sa_m + (long)(k0 + kb)];
                int g  = (kb >> 3) & 1;
                int i0 = ((kb & 16) >> 1) + (kb & 7);   // even
                unsigned* row = AsU[p][m + 16 * g];
                row[i0 >> 1]       = pack2h(va.x, va.y);
                row[(i0 >> 1) + 1] = pack2h(va.z, va.w);
            } else if (a_int && a_colvec) {
                // column-major A (VQ X^T): thread -> 4 consecutive m at fixed k
                int k  = threadIdx.x >> 2;          // 0..31
                int m4 = (threadIdx.x & 3) * 4;     // 0,4,8,12
                float4 va = *(const float4*)&A[(long)(k0 + k) * sa_k + (tm + m4)];
                int g = (k >> 3) & 1;
                int i = ((k & 16) >> 1) + (k & 7);
                float vv[4] = {va.x, va.y, va.z, va.w};
#pragma unroll
                for (int u = 0; u < 4; ++u)
                    ((_Float16*)AsU[p][m4 + u + 16 * g])[i] = (_Float16)vv[u];
            } else {
                int j  = threadIdx.x * 4;
                int m  = j >> 5;
                int kb = j & 31;
#pragma unroll
                for (int u = 0; u < 4; ++u) {
                    int k = kb + u;
                    float v = ((tm + m) < M && (k0 + k) < K)
                            ? A[(long)(tm + m) * sa_m + (long)(k0 + k) * sa_k] : 0.0f;
                    int g = (k >> 3) & 1;
                    int i = ((k & 16) >> 1) + (k & 7);
                    ((_Float16*)AsU[p][m + 16 * g])[i] = (_Float16)v;
                }
            }
        }
        // ---------------- B tile (32k x 32n, this wave) ----------------
        if (b_full_n && (k0 + 32 <= K)) {
            if (b_vec) {
                // float4 along n, pack f16 pairs along k: 8 vec loads, 16 b32 stores
#pragma unroll
                for (int it = 0; it < 4; ++it) {
                    int idx = it * 32 + lane;
                    int kp  = idx >> 3;          // 0..15
                    int n4  = (idx & 7) * 4;     // 0,4,...,28
                    int k   = kp * 2;
                    const float4 r0 = *(const float4*)&B[(long)(k0 + k)     * N + (tn + n4)];
                    const float4 r1 = *(const float4*)&B[(long)(k0 + k + 1) * N + (tn + n4)];
                    float a0[4] = {r0.x, r0.y, r0.z, r0.w};
                    float a1[4] = {r1.x, r1.y, r1.z, r1.w};
#pragma unroll
                    for (int u = 0; u < 4; ++u) {
                        int n = n4 + u;
                        BsU[p][wv][n >> 4][(n & 15) + 16 * ((k >> 3) & 1)][(((k & 16) >> 1) + (k & 7)) >> 1] =
                            pack2h(a0[u], a1[u]);
                    }
                }
            } else {
                // unconditional scalar, pack pairs along k
#pragma unroll
                for (int it = 0; it < 16; ++it) {
                    int idx = it * 32 + lane;
                    int n   = idx & 31;          // lane-consecutive n => coalesced
                    int k   = (idx >> 5) * 2;
                    float v0 = B[(long)(k0 + k)     * N + (tn + n)];
                    float v1 = B[(long)(k0 + k + 1) * N + (tn + n)];
                    BsU[p][wv][n >> 4][(n & 15) + 16 * ((k >> 3) & 1)][(((k & 16) >> 1) + (k & 7)) >> 1] = pack2h(v0, v1);
                }
            }
        } else {
            for (int it = 0; it < 32; ++it) {
                int j = it * 32 + lane;
                int n = j & 31;
                int k = j >> 5;
                float v = ((k0 + k) < K && (tn + n) < N)
                        ? B[(long)(k0 + k) * N + (tn + n)] : 0.0f;
                ((_Float16*)BsU[p][wv][n >> 4][(n & 15) + 16 * ((k >> 3) & 1)])[((k & 16) >> 1) + (k & 7)] = (_Float16)v;
            }
        }
    };

    v8f acc0 = {0.f, 0.f, 0.f, 0.f, 0.f, 0.f, 0.f, 0.f};
    v8f acc1 = {0.f, 0.f, 0.f, 0.f, 0.f, 0.f, 0.f, 0.f};

    stage(0, 0);
    __syncthreads();
    int p = 0;
    for (int k0 = 0; k0 < K; k0 += 32) {
        if (k0 + 32 < K) {
            stage(p ^ 1, k0 + 32);
            if (k0 + 64 < K)
                __builtin_prefetch(&B[(long)(k0 + 64) * N + tn], 0, 1);
        }
        v16h a  = *(const v16h*)AsU[p][lane];
        v16h b0 = *(const v16h*)BsU[p][wv][0][lane];
        v16h b1 = *(const v16h*)BsU[p][wv][1][lane];
        acc0 = __builtin_amdgcn_wmma_f32_16x16x32_f16(false, a, false, b0,
                                                      (short)0, acc0, false, false);
        acc1 = __builtin_amdgcn_wmma_f32_16x16x32_f16(false, a, false, b1,
                                                      (short)0, acc1, false, false);
        __syncthreads();
        p ^= 1;
    }

    // C/D layout: VGPR r -> M = r (lanes 0-15) / r+8 (lanes 16-31), N = lane%16
#pragma unroll
    for (int r = 0; r < 8; ++r) {
        int m = tm + r + half * 8;
        if (m >= M) continue;
        float bv = bias ? bias[m] : 0.0f;
        int n0 = tn + lm;
        if (n0 < N) {
            float v = acc0[r] + bv;
            if (EPI == 1) v = 0.5f * v * (1.0f + erff(v * 0.70710678118654752f));
            if (EPI == 2) v += R[(long)m * N + n0];
            Cmat[(long)m * N + n0] = v;
        }
        int n1 = tn + 16 + lm;
        if (n1 < N) {
            float v = acc1[r] + bv;
            if (EPI == 1) v = 0.5f * v * (1.0f + erff(v * 0.70710678118654752f));
            if (EPI == 2) v += R[(long)m * N + n1];
            Cmat[(long)m * N + n1] = v;
        }
    }
}

// ---------------------------------------------------------------------------
// Bulk device-to-device copy through the Tensor Data Mover (TDM):
// one wave per 32KB chunk: 1-D tensor descriptor, tensor_load_to_lds ->
// s_wait_tensorcnt 0 -> tensor_store_from_lds. Falls back to a plain loop if
// the TDM builtins are unavailable (also covers the host compile pass).
// D# packing per CDNA5 ISA.
// ---------------------------------------------------------------------------
#define TDM_CHUNK 8192  // floats per block (32 KB of LDS)

__global__ __launch_bounds__(32)
void tdm_copy_k(const float* __restrict__ src, float* __restrict__ dst, long n)
{
#if __has_builtin(__builtin_amdgcn_tensor_load_to_lds) && \
    __has_builtin(__builtin_amdgcn_tensor_store_from_lds) && \
    __has_builtin(__builtin_amdgcn_s_wait_tensorcnt)
    __shared__ __align__(16) float lbuf[TDM_CHUNK];
    long base = (long)blockIdx.x * TDM_CHUNK;
    long cnt  = n - base; if (cnt > TDM_CHUNK) cnt = TDM_CHUNK;
    if (cnt <= 0) return;

    unsigned lds_addr = (unsigned)(unsigned long long)(void*)lbuf; // LDS aperture: addr[31:0]
    unsigned long long ga = (unsigned long long)(const void*)(src + base);
    unsigned long long gb = (unsigned long long)(void*)(dst + base);
    unsigned tile = (unsigned)cnt;

    // group0: count=1 | lds_addr | global_addr[56:0] | type=2 (bits 127:126)
    v4u g0l = { 1u, lds_addr, (unsigned)ga,
                (unsigned)((ga >> 32) & 0x01ffffffu) | (2u << 30) };
    v4u g0s = { 1u, lds_addr, (unsigned)gb,
                (unsigned)((gb >> 32) & 0x01ffffffu) | (2u << 30) };
    // group1: data_size=4B; tensor_dim0=tile; tensor_dim1=1; tile_dim0=tile;
    //         tile_dim1/2 unused; tensor_dim0_stride=tile
    v8i g1 = { (int)(2u << 16),
               (int)((tile & 0xffffu) << 16),
               (int)(((tile >> 16) & 0xffffu) | (1u << 16)),
               (int)((tile & 0xffffu) << 16),
               0,
               (int)tile,
               0,
               0 };
    v4i z4 = { 0, 0, 0, 0 };
#if __clang_major__ >= 23
    v8i z8 = { 0, 0, 0, 0, 0, 0, 0, 0 };
    __builtin_amdgcn_tensor_load_to_lds(g0l, g1, z4, z4, z8, 0);
    __builtin_amdgcn_s_wait_tensorcnt(0);
    __builtin_amdgcn_tensor_store_from_lds(g0s, g1, z4, z4, z8, 0);
    __builtin_amdgcn_s_wait_tensorcnt(0);
#else
    __builtin_amdgcn_tensor_load_to_lds(g0l, g1, z4, z4, 0);
    __builtin_amdgcn_s_wait_tensorcnt(0);
    __builtin_amdgcn_tensor_store_from_lds(g0s, g1, z4, z4, 0);
    __builtin_amdgcn_s_wait_tensorcnt(0);
#endif
#else
    long base = (long)blockIdx.x * TDM_CHUNK;
    long end  = base + TDM_CHUNK; if (end > n) end = n;
    for (long i = base + threadIdx.x; i < end; i += 32) dst[i] = src[i];
#endif
}

// ---------------------------------------------------------------------------
// Depthwise 5^3 conv, pad=2, stride in {1,2}
// ---------------------------------------------------------------------------
__global__ __launch_bounds__(256)
void dwconv5_k(const float* __restrict__ x, const float* __restrict__ w,
               const float* __restrict__ bias, float* __restrict__ y,
               int C, int D, int H, int W, int oD, int oH, int oW, int stride)
{
    long oS = (long)oD * oH * oW;
    long idx = (long)blockIdx.x * blockDim.x + threadIdx.x;
    if (idx >= (long)C * oS) return;
    int c = (int)(idx / oS);
    long r = idx % oS;
    int od = (int)(r / ((long)oH * oW));
    int oh = (int)((r / oW) % oH);
    int ow = (int)(r % oW);
    const float* wc = w + (long)c * 125;
    const float* xc = x + (long)c * D * H * W;
    float acc = bias[c];
    for (int kd = 0; kd < 5; ++kd) {
        int id = od * stride + kd - 2; if (id < 0 || id >= D) continue;
        for (int kh = 0; kh < 5; ++kh) {
            int ih = oh * stride + kh - 2; if (ih < 0 || ih >= H) continue;
            for (int kw = 0; kw < 5; ++kw) {
                int iw = ow * stride + kw - 2; if (iw < 0 || iw >= W) continue;
                acc += xc[((long)id * H + ih) * W + iw] * wc[(kd * 5 + kh) * 5 + kw];
            }
        }
    }
    y[idx] = acc;
}

// Depthwise transposed 5^3, stride=2, pad=2  (out = 2*D-1)
__global__ __launch_bounds__(256)
void dwconvT5_k(const float* __restrict__ x, const float* __restrict__ w,
                const float* __restrict__ bias, float* __restrict__ y,
                int C, int D, int oD)
{
    long oS = (long)oD * oD * oD;
    long idx = (long)blockIdx.x * blockDim.x + threadIdx.x;
    if (idx >= (long)C * oS) return;
    int c = (int)(idx / oS);
    long r = idx % oS;
    int od = (int)(r / ((long)oD * oD));
    int oh = (int)((r / oD) % oD);
    int ow = (int)(r % oD);
    const float* wc = w + (long)c * 125;
    const float* xc = x + (long)c * D * D * D;
    float acc = bias[c];
    for (int kd = 0; kd < 5; ++kd) {
        int td = od + 2 - kd; if (td < 0 || (td & 1)) continue;
        int id = td >> 1; if (id >= D) continue;
        for (int kh = 0; kh < 5; ++kh) {
            int th = oh + 2 - kh; if (th < 0 || (th & 1)) continue;
            int ih = th >> 1; if (ih >= D) continue;
            for (int kw = 0; kw < 5; ++kw) {
                int tw = ow + 2 - kw; if (tw < 0 || (tw & 1)) continue;
                int iw = tw >> 1; if (iw >= D) continue;
                acc += xc[((long)id * D + ih) * D + iw] * wc[(kd * 5 + kh) * 5 + kw];
            }
        }
    }
    y[idx] = acc;
}

// Full-channel transposed 1x1x1, stride=2, pad=0 (out = 2*D-1), w[Cin,Cout]
__global__ __launch_bounds__(256)
void convT1s2_k(const float* __restrict__ x, const float* __restrict__ w,
                const float* __restrict__ bias, float* __restrict__ y,
                int Cin, int Cout, int D, int oD)
{
    long oS = (long)oD * oD * oD;
    long idx = (long)blockIdx.x * blockDim.x + threadIdx.x;
    if (idx >= (long)Cout * oS) return;
    int co = (int)(idx / oS);
    long r = idx % oS;
    int od = (int)(r / ((long)oD * oD));
    int oh = (int)((r / oD) % oD);
    int ow = (int)(r % oD);
    float acc = bias[co];
    if (!(od & 1) && !(oh & 1) && !(ow & 1)) {
        long S = (long)D * D * D;
        long xi = ((long)(od >> 1) * D + (oh >> 1)) * D + (ow >> 1);
        for (int ci = 0; ci < Cin; ++ci)
            acc += x[(long)ci * S + xi] * w[(long)ci * Cout + co];
    }
    y[idx] = acc;
}

// Full transposed conv 4^3, stride=2, pad=1 (out = 2*D), w[Cin,Cout,4,4,4]
__global__ __launch_bounds__(256)
void convT4s2_k(const float* __restrict__ x, const float* __restrict__ w,
                const float* __restrict__ bias, float* __restrict__ y,
                int Cin, int Cout, int D, int oD)
{
    long oS = (long)oD * oD * oD;
    long idx = (long)blockIdx.x * blockDim.x + threadIdx.x;
    if (idx >= (long)Cout * oS) return;
    int co = (int)(idx / oS);
    long r = idx % oS;
    int od = (int)(r / ((long)oD * oD));
    int oh = (int)((r / oD) % oD);
    int ow = (int)(r % oD);

    int kd_[2], id_[2], nd = 0;
    int kh_[2], ih_[2], nh = 0;
    int kw_[2], iw_[2], nw = 0;
    for (int k = 0; k < 4; ++k) {
        int t = od + 1 - k;
        if (t >= 0 && !(t & 1) && (t >> 1) < D) { kd_[nd] = k; id_[nd] = t >> 1; ++nd; }
    }
    for (int k = 0; k < 4; ++k) {
        int t = oh + 1 - k;
        if (t >= 0 && !(t & 1) && (t >> 1) < D) { kh_[nh] = k; ih_[nh] = t >> 1; ++nh; }
    }
    for (int k = 0; k < 4; ++k) {
        int t = ow + 1 - k;
        if (t >= 0 && !(t & 1) && (t >> 1) < D) { kw_[nw] = k; iw_[nw] = t >> 1; ++nw; }
    }
    float acc = bias[co];
    long S = (long)D * D * D;
    for (int ci = 0; ci < Cin; ++ci) {
        const float* xc = x + (long)ci * S;
        const float* wc = w + ((long)ci * Cout + co) * 64;
        for (int a = 0; a < nd; ++a)
            for (int b = 0; b < nh; ++b)
                for (int c2 = 0; c2 < nw; ++c2)
                    acc += xc[((long)id_[a] * D + ih_[b]) * D + iw_[c2]] *
                           wc[kd_[a] * 16 + kh_[b] * 4 + kw_[c2]];
    }
    y[idx] = acc;
}

// Dense 3^3 conv, pad=1, stride=1, w[Cout,Cin,3,3,3]
__global__ __launch_bounds__(256)
void conv3_k(const float* __restrict__ x, const float* __restrict__ w,
             const float* __restrict__ bias, float* __restrict__ y,
             int Cin, int Cout, int D)
{
    long S = (long)D * D * D;
    long idx = (long)blockIdx.x * blockDim.x + threadIdx.x;
    if (idx >= (long)Cout * S) return;
    int co = (int)(idx / S);
    long r = idx % S;
    int od = (int)(r / ((long)D * D));
    int oh = (int)((r / D) % D);
    int ow = (int)(r % D);
    float acc = bias[co];
    for (int ci = 0; ci < Cin; ++ci) {
        const float* xc = x + (long)ci * S;
        const float* wc = w + ((long)co * Cin + ci) * 27;
        for (int kd = 0; kd < 3; ++kd) {
            int id = od + kd - 1; if (id < 0 || id >= D) continue;
            for (int kh = 0; kh < 3; ++kh) {
                int ih = oh + kh - 1; if (ih < 0 || ih >= D) continue;
                for (int kw = 0; kw < 3; ++kw) {
                    int iw = ow + kw - 1; if (iw < 0 || iw >= D) continue;
                    acc += xc[((long)id * D + ih) * D + iw] * wc[(kd * 3 + kh) * 3 + kw];
                }
            }
        }
    }
    y[idx] = acc;
}

// Per-channel GroupNorm (num_groups == C), B=1. In-place safe.
__global__ __launch_bounds__(256)
void group_norm_k(const float* __restrict__ x, const float* __restrict__ w,
                  const float* __restrict__ b, float* __restrict__ y, long S)
{
    int c = blockIdx.x;
    const float* xc = x + (long)c * S;
    float* yc = y + (long)c * S;
    float s = 0.f, s2 = 0.f;
    for (long i = threadIdx.x; i < S; i += 256) { float v = xc[i]; s += v; s2 += v * v; }
    __shared__ float rs[256], rq[256];
    rs[threadIdx.x] = s; rq[threadIdx.x] = s2; __syncthreads();
    for (int o = 128; o > 0; o >>= 1) {
        if (threadIdx.x < o) { rs[threadIdx.x] += rs[threadIdx.x + o]; rq[threadIdx.x] += rq[threadIdx.x + o]; }
        __syncthreads();
    }
    float mu  = rs[0] / (float)S;
    float var = rq[0] / (float)S - mu * mu;
    float inv = rsqrtf(var + 1e-5f);
    float ww = w[c], bb = b[c];
    for (long i = threadIdx.x; i < S; i += 256) yc[i] = (xc[i] - mu) * inv * ww + bb;
}

// GRN pieces
__global__ __launch_bounds__(256)
void grn_sumsq_k(const float* __restrict__ x, float* __restrict__ gx, long S)
{
    int c = blockIdx.x;
    const float* xc = x + (long)c * S;
    float s = 0.f;
    for (long i = threadIdx.x; i < S; i += 256) { float v = xc[i]; s += v * v; }
    __shared__ float rs[256];
    rs[threadIdx.x] = s; __syncthreads();
    for (int o = 128; o > 0; o >>= 1) {
        if (threadIdx.x < o) rs[threadIdx.x] += rs[threadIdx.x + o];
        __syncthreads();
    }
    if (threadIdx.x == 0) gx[c] = sqrtf(rs[0]);
}

__global__ __launch_bounds__(512)
void grn_scale_k(const float* __restrict__ gx, const float* __restrict__ gamma,
                 float* __restrict__ scale, int C)
{
    __shared__ float rs[512];
    int t = threadIdx.x;
    rs[t] = (t < C) ? gx[t] : 0.0f;
    __syncthreads();
    for (int o = 256; o > 0; o >>= 1) {
        if (t < o) rs[t] += rs[t + o];
        __syncthreads();
    }
    float mean = rs[0] / (float)C;
    if (t < C) scale[t] = gamma[t] * gx[t] / (mean + 1e-6f) + 1.0f;  // y = x*scale + beta
}

__global__ __launch_bounds__(256)
void grn_apply_k(float* __restrict__ x, const float* __restrict__ scale,
                 const float* __restrict__ beta, long S, long total)
{
    long i = (long)blockIdx.x * blockDim.x + threadIdx.x;
    if (i < total) { long c = i / S; x[i] = x[i] * scale[c] + beta[c]; }
}

// utility kernels
__global__ __launch_bounds__(256)
void subsample2_k(const float* __restrict__ x, float* __restrict__ y, int C, int D, int oD)
{
    long oS = (long)oD * oD * oD;
    long idx = (long)blockIdx.x * blockDim.x + threadIdx.x;
    if (idx >= (long)C * oS) return;
    int c = (int)(idx / oS);
    long r = idx % oS;
    int od = (int)(r / ((long)oD * oD));
    int oh = (int)((r / oD) % oD);
    int ow = (int)(r % oD);
    y[idx] = x[(((long)c * D + od * 2) * D + oh * 2) * D + ow * 2];
}

__global__ __launch_bounds__(256)
void add_padfront_k(const float* __restrict__ a, const float* __restrict__ b,
                    float* __restrict__ y, int C, int iD, int oD)
{
    long oS = (long)oD * oD * oD;
    long idx = (long)blockIdx.x * blockDim.x + threadIdx.x;
    if (idx >= (long)C * oS) return;
    int c = (int)(idx / oS);
    long r = idx % oS;
    int od = (int)(r / ((long)oD * oD));
    int oh = (int)((r / oD) % oD);
    int ow = (int)(r % oD);
    float v = 0.0f;
    if (od > 0 && oh > 0 && ow > 0) {
        long ii = (((long)c * iD + (od - 1)) * iD + (oh - 1)) * iD + (ow - 1);
        v = a[ii] + b[ii];
    }
    y[idx] = v;
}

__global__ void zero_k(float* p, long n)
{
    long i = (long)blockIdx.x * blockDim.x + threadIdx.x;
    if (i < n) p[i] = 0.0f;
}

// VQ pieces
__global__ __launch_bounds__(256)
void sqnorm_cols_k(const float* __restrict__ E, float* __restrict__ en, int C, int NE)
{
    int e = blockIdx.x * blockDim.x + threadIdx.x;
    if (e >= NE) return;
    float s = 0.f;
    for (int c = 0; c < C; ++c) { float v = E[(long)c * NE + e]; s += v * v; }
    en[e] = s;
}

__global__ __launch_bounds__(256)
void argmin_k(const float* __restrict__ G, const float* __restrict__ en,
              int* __restrict__ ind, long S, int NE)
{
    long s = (long)blockIdx.x * blockDim.x + threadIdx.x;
    if (s >= S) return;
    float best = 3.4e38f; int bi = 0;
    for (int e = 0; e < NE; ++e) {
        float d = en[e] - 2.0f * G[s * NE + e];
        if (d < best) { best = d; bi = e; }
    }
    ind[s] = bi;
}

__global__ __launch_bounds__(256)
void gather_k(const float* __restrict__ E, const int* __restrict__ ind,
              float* __restrict__ q, int NE, long S, long total)
{
    long i = (long)blockIdx.x * blockDim.x + threadIdx.x;
    if (i >= total) return;
    long c = i / S, s = i % S;
    q[i] = E[c * NE + ind[s]];
}

__global__ __launch_bounds__(256)
void diffsum_k(const float* __restrict__ q, const float* __restrict__ x,
               float* __restrict__ sum, long total)
{
    float s = 0.f;
    for (long i = (long)blockIdx.x * blockDim.x + threadIdx.x; i < total;
         i += (long)gridDim.x * blockDim.x) {
        float d = q[i] - x[i]; s += d * d;
    }
    __shared__ float rs[256];
    rs[threadIdx.x] = s; __syncthreads();
    for (int o = 128; o > 0; o >>= 1) {
        if (threadIdx.x < o) rs[threadIdx.x] += rs[threadIdx.x + o];
        __syncthreads();
    }
    if (threadIdx.x == 0) atomicAdd(sum, rs[0]);
}

__global__ void finalize_k(float* out, const float* sums, float inv0, float inv1)
{
    if (threadIdx.x == 0 && blockIdx.x == 0)
        out[0] = sums[0] * inv0 + sums[1] * inv1;
}

// ---------------------------------------------------------------------------
// Host orchestration
// ---------------------------------------------------------------------------
struct MB {
    const float *dw_w, *dw_b, *norm_w, *norm_b, *pw1_w, *pw1_b, *pw2_w, *pw2_b,
                *gamma, *beta, *res_w, *res_b;
};

extern "C" void kernel_launch(void* const* d_in, const int* in_sizes, int n_in,
                              void* d_out, int out_size, void* d_ws, size_t ws_size,
                              hipStream_t stream)
{
    (void)in_sizes; (void)n_in; (void)out_size; (void)ws_size;

    // ---- unpack inputs in setup_inputs() dict order ----
    int ip = 0;
    auto nxt = [&]() { return (const float*)d_in[ip++]; };
    auto load_mb = [&](bool res) {
        MB m;
        m.dw_w = nxt(); m.dw_b = nxt(); m.norm_w = nxt(); m.norm_b = nxt();
        m.pw1_w = nxt(); m.pw1_b = nxt(); m.pw2_w = nxt(); m.pw2_b = nxt();
        m.gamma = nxt(); m.beta = nxt();
        if (res) { m.res_w = nxt(); m.res_b = nxt(); }
        else     { m.res_w = nullptr; m.res_b = nullptr; }
        return m;
    };

    const float* x_in   = nxt();
    const float* stem_w = nxt(); const float* stem_b = nxt();
    MB eb[2]; for (int i = 0; i < 2; ++i) eb[i] = load_mb(false);
    MB eb_down = load_mb(true);
    MB et[2]; for (int i = 0; i < 2; ++i) et[i] = load_mb(false);
    MB et_down = load_mb(true);
    const float* qconv_t_w = nxt(); const float* qconv_t_b = nxt();
    const float* embed_t   = nxt();
    MB dt[2]; for (int i = 0; i < 2; ++i) dt[i] = load_mb(false);
    MB dt_up = load_mb(true);
    const float* qconv_b_w = nxt(); const float* qconv_b_b = nxt();
    const float* embed_b   = nxt();
    const float* ups_w = nxt(); const float* ups_b = nxt();
    MB db[2]; for (int i = 0; i < 2; ++i) db[i] = load_mb(false);
    MB d_up = load_mb(true);
    const float* outc_w = nxt(); const float* outc_b = nxt();

    float* out = (float*)d_out;

    // ---- dims ----
    const int  D64 = 64, D32 = 32, D16 = 16;
    const long S64 = 262144, S32 = 32768, S31 = 29791, S16 = 4096, S63 = 250047;

    // ---- workspace bump allocator ----
    size_t off = 0;
    auto alloc = [&](long nf) -> float* {
        float* p = (float*)((char*)d_ws + off);
        off += ((size_t)nf * sizeof(float) + 255) & ~(size_t)255;
        return p;
    };

    // persistent
    float* enc_b   = alloc(64 * S32);
    float* quant_t = alloc(128 * S16);
    float* quant_b = alloc(128 * S32);
    float* gx      = alloc(512);
    float* scalev  = alloc(512);
    float* env     = alloc(512);
    float* sums    = alloc(2);
    int*   ind_t   = (int*)alloc(4096);
    int*   ind_b   = (int*)alloc(32768);
    const size_t mark = off;

    auto blks = [](long n) { return (unsigned)((n + 255) / 256); };

    auto gemm = [&](int epi, const float* A, long sam, long sak, const float* B,
                    const float* bias, const float* R, float* Cp, int M, int N, int K) {
        dim3 g((unsigned)((N + 127) / 128), (unsigned)((M + 15) / 16));
        if (epi == 0)      gemm_wmma_k<0><<<g, 128, 0, stream>>>(A, sam, sak, B, bias, R, Cp, M, N, K);
        else if (epi == 1) gemm_wmma_k<1><<<g, 128, 0, stream>>>(A, sam, sak, B, bias, R, Cp, M, N, K);
        else               gemm_wmma_k<2><<<g, 128, 0, stream>>>(A, sam, sak, B, bias, R, Cp, M, N, K);
    };

    auto tdm_copy = [&](const float* src, float* dst, long n) {
        unsigned nb = (unsigned)((n + TDM_CHUNK - 1) / TDM_CHUNK);
        tdm_copy_k<<<nb, 32, 0, stream>>>(src, dst, n);
    };

    auto grn = [&](float* t, const float* gamma, const float* beta, int Cc, long S) {
        grn_sumsq_k<<<Cc, 256, 0, stream>>>(t, gx, S);
        grn_scale_k<<<1, 512, 0, stream>>>(gx, gamma, scalev, Cc);
        grn_apply_k<<<blks((long)Cc * S), 256, 0, stream>>>(t, scalev, beta, S, (long)Cc * S);
    };

    // mednext residual block: out = x + pw2(grn(gelu(pw1(gn(dw5(x))))))
    auto mednext = [&](const MB& p, const float* xin, float* t1, float* t2, float* outp,
                       int C, int mid, int D) {
        long S = (long)D * D * D;
        dwconv5_k<<<blks((long)C * S), 256, 0, stream>>>(xin, p.dw_w, p.dw_b, t1,
                                                         C, D, D, D, D, D, D, 1);
        group_norm_k<<<C, 256, 0, stream>>>(t1, p.norm_w, p.norm_b, t1, S);
        gemm(1, p.pw1_w, C, 1, t1, p.pw1_b, nullptr, t2, mid, (int)S, C);
        grn(t2, p.gamma, p.beta, mid, S);
        gemm(2, p.pw2_w, mid, 1, t2, p.pw2_b, xin, outp, C, (int)S, mid);
    };

    // down block: out = block_core(x, stride2) + res_conv(x, stride2)
    auto down = [&](const MB& p, const float* xin, float* t1, float* t2, float* x1,
                    float* xs, float* outp, int Cin, int Cout, int mid, int D) {
        int oD = D / 2; long S = (long)oD * oD * oD;
        dwconv5_k<<<blks((long)Cin * S), 256, 0, stream>>>(xin, p.dw_w, p.dw_b, t1,
                                                           Cin, D, D, D, oD, oD, oD, 2);
        group_norm_k<<<Cin, 256, 0, stream>>>(t1, p.norm_w, p.norm_b, t1, S);
        gemm(1, p.pw1_w, Cin, 1, t1, p.pw1_b, nullptr, t2, mid, (int)S, Cin);
        grn(t2, p.gamma, p.beta, mid, S);
        gemm(0, p.pw2_w, mid, 1, t2, p.pw2_b, nullptr, x1, Cout, (int)S, mid);
        subsample2_k<<<blks((long)Cin * S), 256, 0, stream>>>(xin, xs, Cin, D, oD);
        gemm(2, p.res_w, Cin, 1, xs, p.res_b, x1, outp, Cout, (int)S, Cin);
    };

    // up block: out = pad_front(block_core_T(x)) + pad_front(res_convT(x))
    auto up = [&](const MB& p, const float* xin, float* t1, float* t2, float* x1,
                  float* res, float* outp, int Cin, int Cout, int mid, int D) {
        int mD = 2 * D - 1; long Sm = (long)mD * mD * mD;
        int oD = 2 * D;     long So = (long)oD * oD * oD;
        dwconvT5_k<<<blks((long)Cin * Sm), 256, 0, stream>>>(xin, p.dw_w, p.dw_b, t1, Cin, D, mD);
        group_norm_k<<<Cin, 256, 0, stream>>>(t1, p.norm_w, p.norm_b, t1, Sm);
        gemm(1, p.pw1_w, Cin, 1, t1, p.pw1_b, nullptr, t2, mid, (int)Sm, Cin);
        grn(t2, p.gamma, p.beta, mid, Sm);
        gemm(0, p.pw2_w, mid, 1, t2, p.pw2_b, nullptr, x1, Cout, (int)Sm, mid);
        convT1s2_k<<<blks((long)Cout * Sm), 256, 0, stream>>>(xin, p.res_w, p.res_b, res,
                                                              Cin, Cout, D, mD);
        add_padfront_k<<<blks((long)Cout * So), 256, 0, stream>>>(x1, res, outp, Cout, mD, oD);
    };

    // VQ: dist argmin via GEMM G[s,e] = sum_c X[c,s]*E[c,e]; argmin(en - 2G)
    auto vq = [&](const float* xq, const float* E, float* G, int* ind, float* q,
                  float* sumslot, long S) {
        gemm(0, xq, 1, S, E, nullptr, nullptr, G, (int)S, 512, 128);
        sqnorm_cols_k<<<2, 256, 0, stream>>>(E, env, 128, 512);
        argmin_k<<<blks(S), 256, 0, stream>>>(G, env, ind, S, 512);
        gather_k<<<blks(128 * S), 256, 0, stream>>>(E, ind, q, 512, S, 128 * S);
        diffsum_k<<<blks(128 * S), 256, 0, stream>>>(q, xq, sumslot, 128 * S);
    };

    zero_k<<<1, 32, 0, stream>>>(sums, 2);

    // ================= Phase 1: encoder bottom (64^3, C=32) =================
    off = mark;
    float* A0 = alloc(32 * S64);
    float* A1 = alloc(32 * S64);
    float* A2 = alloc(32 * S64);
    float* B1 = alloc(32 * S32);
    float* B2 = alloc(64 * S32);
    float* B3 = alloc(64 * S32);
    float* B4 = alloc(32 * S32);

    conv3_k<<<blks(32 * S64), 256, 0, stream>>>(x_in, stem_w, stem_b, A0, 1, 32, D64);
    mednext(eb[0], A0, A1, A2, A1, 32, 32, D64);
    mednext(eb[1], A1, A2, A0, A2, 32, 32, D64);
    down(eb_down, A2, B1, B2, B3, B4, enc_b, 32, 64, 64, D64);

    // ================= Phase 2: encoder top (32^3, C=64) + VQ_t =============
    off = mark;
    float* C0 = alloc(64 * S32);
    float* C1 = alloc(64 * S32);
    float* C2 = alloc(64 * S32);
    float* D1 = alloc(64 * S16);
    float* D2 = alloc(128 * S16);
    float* D3 = alloc(128 * S16);
    float* D4 = alloc(64 * S16);
    float* D5 = alloc(128 * S16);   // enc_t
    float* D6 = alloc(128 * S16);   // qt
    float* D7 = alloc(4096 * 512);  // G_t

    mednext(et[0], enc_b, C1, C2, C0, 64, 64, D32);
    mednext(et[1], C0, C1, C2, C1, 64, 64, D32);
    down(et_down, C1, D1, D2, D3, D4, D5, 64, 128, 128, D32);
    gemm(0, qconv_t_w, 128, 1, D5, qconv_t_b, nullptr, D6, 128, (int)S16, 128);
    vq(D6, embed_t, D7, ind_t, quant_t, &sums[0], S16);

    // ================= Phase 3: decoder top (16^3, C=128) + VQ_b ============
    off = mark;
    float* E0 = alloc(128 * S16);
    float* E1 = alloc(128 * S16);
    float* E2 = alloc(256 * S16);
    float* F1 = alloc(128 * S31);
    float* F2 = alloc(256 * S31);
    float* F3 = alloc(64 * S31);
    float* F4 = alloc(64 * S31);
    float* F5 = alloc(64 * S32);    // dec_t
    float* G0 = alloc(128 * S32);   // cat_b
    float* G1 = alloc(128 * S32);   // qb
    float* G2 = alloc(32768L * 512);// G_b

    mednext(dt[0], quant_t, E1, E2, E0, 128, 256, D16);
    mednext(dt[1], E0, E1, E2, E1, 128, 256, D16);
    up(dt_up, E1, F1, F2, F3, F4, F5, 128, 64, 256, D16);
    tdm_copy(F5, G0, 64 * S32);
    tdm_copy(enc_b, G0 + 64 * S32, 64 * S32);
    gemm(0, qconv_b_w, 128, 1, G0, qconv_b_b, nullptr, G1, 128, (int)S32, 128);
    vq(G1, embed_b, G2, ind_b, quant_b, &sums[1], S32);

    // ================= Phase 4: decoder bottom (32^3 -> 64^3) ===============
    off = mark;
    float* H0 = alloc(256 * S32);
    float* H1 = alloc(256 * S32);
    float* H2 = alloc(512 * S32);
    float* K1 = alloc(256 * S63);
    float* K2 = alloc(512 * S63);
    float* K3 = alloc(128 * S63);
    float* K4 = alloc(128 * S63);
    float* K5 = alloc(128 * S64);

    // h = concat(upsample_t(quant_t), quant_b)
    convT4s2_k<<<blks(128 * S32), 256, 0, stream>>>(quant_t, ups_w, ups_b, H0, 128, 128, D16, D32);
    tdm_copy(quant_b, H0 + 128 * S32, 128 * S32);

    mednext(db[0], H0, H1, H2, H1, 256, 512, D32);
    mednext(db[1], H1, H0, H2, H0, 256, 512, D32);
    up(d_up, H0, K1, K2, K3, K4, K5, 256, 128, 512, D32);

    conv3_k<<<blks(1 * S64), 256, 0, stream>>>(K5, outc_w, outc_b, out, 128, 1, D64);
    finalize_k<<<1, 32, 0, stream>>>(out + S64, sums,
                                     1.0f / (float)(128 * S16), 1.0f / (float)(128 * S32));
}
